// EdgeFusionGATLayer_45947560133205
// MI455X (gfx1250) — matrix operand order repaired and checked by reading, not compiled, not run on hardware
//
#include <hip/hip_runtime.h>

#define IN   128
#define OUT  128
#define NH   4
#define HD   32
#define ED   32

typedef __attribute__((ext_vector_type(16))) __bf16 v16bf;
typedef __attribute__((ext_vector_type(8)))  __bf16 v8bf;
typedef __attribute__((ext_vector_type(8)))  float  v8f;
typedef unsigned int __attribute__((ext_vector_type(4))) v4u;
typedef int __attribute__((ext_vector_type(8))) v8i;
typedef int __attribute__((ext_vector_type(4))) v4i;

// ---------------- WMMA helpers (CDNA5 16x16x32 bf16, f32 accum) ----------------

__device__ inline v8f wmma_bf16(v16bf a, v16bf b, v8f c) {
  return __builtin_amdgcn_wmma_f32_16x16x32_bf16(false, a, false, b, (short)0, c,
                                                 false, false);
}

// A fragment: 16x32 bf16 from LDS row [k0..k0+31]. Lanes 0-15: K {0..7,16..23};
// lanes 16-31: K {8..15,24..31} (per ISA 16-bit A layout).
__device__ inline v16bf load_frag_a(const __bf16* base, int lane) {
  int kb = (lane < 16) ? 0 : 8;
  v8bf lo = *(const v8bf*)(base + kb);
  v8bf hi = *(const v8bf*)(base + kb + 16);
  v16bf r;
#pragma unroll
  for (int i = 0; i < 8; ++i) { r[i] = lo[i]; r[i + 8] = hi[i]; }
  return r;
}

// B fragment: 32x16 bf16 from W^T stored [col][K]. Lanes 0-15: K {0..7,16..23};
// lanes 16-31: K {8..15,24..31}.
__device__ inline v16bf load_frag_b(const __bf16* base, int lane) {
  int kb = (lane < 16) ? 0 : 8;
  v8bf lo = *(const v8bf*)(base + kb);
  v8bf hi = *(const v8bf*)(base + kb + 16);
  v16bf r;
#pragma unroll
  for (int i = 0; i < 8; ++i) { r[i] = lo[i]; r[i + 8] = hi[i]; }
  return r;
}

// ---------------- TDM: 2D tensor tile -> LDS (bf16 elements) ----------------
// Builds D# group0/group1 per CDNA5 ISA 8.3/8.4 and issues tensor_load_to_lds.
// tile_w/tile_h: tile dims (elements/rows); stride: row stride in elements;
// tensor_w/tensor_h: valid extent for OOB zero-fill.
__device__ inline void tdm_load_2d_bf16(unsigned lds_addr, const void* gptr,
                                        unsigned tile_w, unsigned tile_h,
                                        unsigned long long stride_elems,
                                        unsigned tensor_w, unsigned tensor_h) {
  unsigned long long ga = (unsigned long long)gptr;
  v4u g0;
  g0[0] = 1u;                                           // count=1 (valid), load
  g0[1] = lds_addr;                                     // lds_addr [63:32]
  g0[2] = (unsigned)(ga & 0xFFFFFFFFull);               // global_addr lo
  g0[3] = (unsigned)((ga >> 32) & 0x01FFFFFFull)        // global_addr hi (57b)
          | (2u << 30);                                 // type=2 ("image")
  v8i g1;
  g1[0] = 0x00010000;                                   // data_size=1 (2 bytes)
  g1[1] = (int)((tensor_w & 0xFFFFu) << 16);            // tensor_dim0 [79:48] lo
  g1[2] = (int)(((tensor_w >> 16) & 0xFFFFu) |
                ((tensor_h & 0xFFFFu) << 16));          // dim0 hi | dim1 lo
  g1[3] = (int)(((tensor_h >> 16) & 0xFFFFu) |
                ((tile_w & 0xFFFFu) << 16));            // dim1 hi | tile_dim0
  g1[4] = (int)(tile_h & 0xFFFFu);                      // tile_dim1 (tile_dim2=0)
  g1[5] = (int)(stride_elems & 0xFFFFFFFFull);          // dim0_stride lo32
  g1[6] = (int)((stride_elems >> 32) & 0xFFFFull);      // dim0_stride hi16
  g1[7] = 0;
  v4i z4 = {0, 0, 0, 0};
  v8i z8 = {0, 0, 0, 0, 0, 0, 0, 0};
  __builtin_amdgcn_tensor_load_to_lds(g0, g1, z4, z4, z8, 0);
}

// monotone float<->uint mapping for atomicMax on signed floats
__device__ inline unsigned enc_f32(float f) {
  unsigned u = __float_as_uint(f);
  return (u & 0x80000000u) ? ~u : (u | 0x80000000u);
}
__device__ inline float dec_f32(unsigned u) {
  unsigned b = (u & 0x80000000u) ? (u & 0x7FFFFFFFu) : ~u;
  return __uint_as_float(b);
}

// ---------------- prep kernels ----------------

__global__ void zero_kernel(float* p, long n) {
  long i = (long)blockIdx.x * 256 + threadIdx.x;
  if (i < n) p[i] = 0.0f;
}

__global__ void to_bf16_kernel(const float* __restrict__ x, __bf16* __restrict__ y, long n) {
  long i = (long)blockIdx.x * 256 + threadIdx.x;
  if (i < n) y[i] = (__bf16)x[i];
}

// Wt[c*K + k] = W[k*C + c]   (W is [K,C] row-major)
__global__ void transpose_bf16_kernel(const float* __restrict__ W, __bf16* __restrict__ Wt,
                                      int K, int C) {
  long i = (long)blockIdx.x * 256 + threadIdx.x;
  if (i >= (long)K * C) return;
  int c = (int)(i / K), k = (int)(i % K);
  Wt[(long)c * K + k] = (__bf16)W[(long)k * C + c];
}

// ---------------- GEMM kernels ----------------

// feat_src = node @ W_src + b_src ; feat_dst likewise. 32 rows per block; TDM staging.
__global__ __launch_bounds__(256)
void node_linear_kernel(const __bf16* __restrict__ node_bf,
                        const __bf16* __restrict__ WsrcT, const float* __restrict__ b_src,
                        const __bf16* __restrict__ WdstT, const float* __restrict__ b_dst,
                        float* __restrict__ feat_src, float* __restrict__ feat_dst, int N) {
  __shared__ __bf16 As[32][IN];      // 8 KB  (TDM tile: 32 rows x 128)
  __shared__ __bf16 Wt[OUT][IN];     // 32 KB (TDM tile: 128 cols x 128 K)
  int tid = threadIdx.x;
  long n0 = (long)blockIdx.x * 32;
  int wave = tid >> 5, lane = tid & 31;
  int rt = wave >> 2;  // row tile 0/1

  if (wave == 0) {
    unsigned remA = (unsigned)(N - n0);  // OOB rows read as zero
    tdm_load_2d_bf16((unsigned)(unsigned long long)&As[0][0], node_bf + n0 * IN,
                     IN, 32, IN, IN, remA);
  }

  for (int mat = 0; mat < 2; ++mat) {
    const __bf16* WT = mat == 0 ? WsrcT : WdstT;
    const float* bias = mat == 0 ? b_src : b_dst;
    float* outp = mat == 0 ? feat_src : feat_dst;
    __syncthreads();  // previous iter's Wt reads done
    if (wave == 0) {
      tdm_load_2d_bf16((unsigned)(unsigned long long)&Wt[0][0], WT, IN, OUT, IN, IN, OUT);
      __builtin_amdgcn_s_wait_tensorcnt(0);
    }
    __syncthreads();
#pragma unroll
    for (int sub = 0; sub < 2; ++sub) {
      int ct = (wave & 3) + sub * 4;  // col tile 0..7
      v8f acc = {};
#pragma unroll
      for (int k0 = 0; k0 < IN; k0 += 32) {
        v16bf a = load_frag_a(&As[rt * 16 + (lane & 15)][k0], lane);
        v16bf b = load_frag_b(&Wt[ct * 16 + (lane & 15)][k0], lane);
        acc = wmma_bf16(a, b, acc);
      }
      int colg = ct * 16 + (lane & 15);
      float bv = bias[colg];
      long nb = n0 + rt * 16 + ((lane >> 4) << 3);
      if (n0 + 32 <= N) {
#pragma unroll
        for (int j = 0; j < 8; ++j) outp[(nb + j) * OUT + colg] = acc[j] + bv;
      } else {
#pragma unroll
        for (int j = 0; j < 8; ++j) {
          long n = nb + j;
          if (n < N) outp[n * OUT + colg] = acc[j] + bv;
        }
      }
    }
  }
}

// per (edge, head): score = a_h . leaky_relu(fs + fd); atomic segment max
__global__ __launch_bounds__(256)
void edge_score_kernel(const float* __restrict__ feat_src, const float* __restrict__ feat_dst,
                       const float* __restrict__ attn,
                       const int* __restrict__ src, const int* __restrict__ dst,
                       float* __restrict__ score, unsigned* __restrict__ smax, long EH) {
  int lane = threadIdx.x & 31;
  long gw = (long)blockIdx.x * 8 + (threadIdx.x >> 5);
  if (gw >= EH) return;
  long e = gw >> 2;
  int h = (int)(gw & 3);
  int s = src[e], d = dst[e];
  float fs = feat_src[(long)s * OUT + h * HD + lane];
  float fd = feat_dst[(long)d * OUT + h * HD + lane];
  float v = fs + fd;
  v = v > 0.0f ? v : 0.2f * v;
  float p = v * attn[h * HD + lane];
#pragma unroll
  for (int m = 16; m > 0; m >>= 1) p += __shfl_xor(p, m, 32);
  if (lane == 0) {
    score[gw] = p;
    atomicMax(&smax[(long)d * NH + h], enc_f32(p));
  }
}

// ex = exp(score - smax[dst]); denom += ex (score buffer overwritten with ex)
__global__ __launch_bounds__(256)
void edge_exp_kernel(const int* __restrict__ dst, const unsigned* __restrict__ smax,
                     float* __restrict__ score, float* __restrict__ denom, long EH) {
  long i = (long)blockIdx.x * 256 + threadIdx.x;
  if (i >= EH) return;
  long e = i >> 2;
  int h = (int)(i & 3);
  int d = dst[e];
  float m = dec_f32(smax[(long)d * NH + h]);
  float x = __expf(score[i] - m);
  score[i] = x;
  atomicAdd(&denom[(long)d * NH + h], x);
}

// alpha = ex/denom; h_gat[dst] += alpha * feat_src[src]; write alpha output
__global__ __launch_bounds__(256)
void edge_aggregate_kernel(const float* __restrict__ feat_src,
                           const int* __restrict__ src, const int* __restrict__ dst,
                           const float* __restrict__ ex, const float* __restrict__ denom,
                           float* __restrict__ h_gat, float* __restrict__ out_alpha, long E) {
  int lane = threadIdx.x & 31;
  long e = (long)blockIdx.x * 8 + (threadIdx.x >> 5);
  if (e >= E) return;
  int s = src[e], d = dst[e];
#pragma unroll
  for (int h = 0; h < NH; ++h) {
    int c = h * 32 + lane;
    float al = ex[e * NH + h] / denom[(long)d * NH + h];
    atomicAdd(&h_gat[(long)d * OUT + c], al * feat_src[(long)s * OUT + c]);
  }
  if (lane < NH)
    out_alpha[e * NH + lane] = ex[e * NH + lane] / denom[(long)d * NH + lane];
}

// edge_emb = elu(LN(edge_feat @ We + be)); edge_agg[dst] += edge_emb; cnt[dst] += 1
__global__ __launch_bounds__(256)
void edge_emb_kernel(const __bf16* __restrict__ edge_bf,
                     const __bf16* __restrict__ WeTg, const float* __restrict__ be,
                     const float* __restrict__ g, const float* __restrict__ bb,
                     const int* __restrict__ dst,
                     float* __restrict__ edge_agg, float* __restrict__ cnt, long E) {
  __shared__ __bf16 WeT[OUT][ED];    // 8 KB (TDM tile: 128 cols x 32 K)
  __shared__ __bf16 As[16][ED];      // 1 KB (TDM tile: 16 edges x 32)
  __shared__ float lnbuf[16][OUT];   // 8 KB
  int tid = threadIdx.x;
  long e0 = (long)blockIdx.x * 16;
  int wave = tid >> 5, lane = tid & 31;

  if (wave == 0) {
    tdm_load_2d_bf16((unsigned)(unsigned long long)&As[0][0], edge_bf + e0 * ED,
                     ED, 16, ED, ED, (unsigned)(E - e0));
    tdm_load_2d_bf16((unsigned)(unsigned long long)&WeT[0][0], WeTg, ED, OUT, ED, ED, OUT);
    __builtin_amdgcn_s_wait_tensorcnt(0);
  }
  __syncthreads();

  int ct = wave;  // 8 col tiles
  v16bf a = load_frag_a(&As[lane & 15][0], lane);
  v16bf b = load_frag_b(&WeT[ct * 16 + (lane & 15)][0], lane);
  v8f acc = {};
  acc = wmma_bf16(a, b, acc);
  int colg = ct * 16 + (lane & 15);
  float bv = be[colg];
#pragma unroll
  for (int j = 0; j < 8; ++j)
    lnbuf[j + ((lane >> 4) << 3)][colg] = acc[j] + bv;
  __syncthreads();

  int er = tid >> 4, li = tid & 15;
  float s1 = 0.0f;
#pragma unroll
  for (int j = 0; j < 8; ++j) s1 += lnbuf[er][li + 16 * j];
#pragma unroll
  for (int m = 1; m < 16; m <<= 1) s1 += __shfl_xor(s1, m, 16);
  float mean = s1 * (1.0f / 128.0f);
  float s2 = 0.0f;
#pragma unroll
  for (int j = 0; j < 8; ++j) {
    float dq = lnbuf[er][li + 16 * j] - mean;
    s2 += dq * dq;
  }
#pragma unroll
  for (int m = 1; m < 16; m <<= 1) s2 += __shfl_xor(s2, m, 16);
  float rstd = rsqrtf(s2 * (1.0f / 128.0f) + 1e-5f);

  long e = e0 + er;
  if (e < E) {
    int dn = dst[e];
#pragma unroll
    for (int j = 0; j < 8; ++j) {
      int c = li + 16 * j;
      float y = (lnbuf[er][c] - mean) * rstd * g[c] + bb[c];
      y = y > 0.0f ? y : expm1f(y);
      atomicAdd(&edge_agg[(long)dn * OUT + c], y);
    }
    if (li == 0) atomicAdd(&cnt[dn], 1.0f);
  }
}

// out = elu(LN(concat(h_gat, edge_agg/cnt) @ Wf + bf)) ; 16 nodes per block
__global__ __launch_bounds__(256)
void final_kernel(const float* __restrict__ h_gat, const float* __restrict__ edge_agg,
                  const float* __restrict__ cnt,
                  const __bf16* __restrict__ WfTg, const float* __restrict__ bf_,
                  const float* __restrict__ g, const float* __restrict__ bb,
                  float* __restrict__ out, int N) {
  __shared__ __bf16 Wt[OUT][128];    // 32 KB (one K-half of Wf^T via TDM)
  __shared__ __bf16 As[16][128];     // 4 KB
  __shared__ float lnbuf[16][OUT];   // 8 KB
  int tid = threadIdx.x;
  long n0 = (long)blockIdx.x * 16;
  int wave = tid >> 5, lane = tid & 31;
  int ct = wave;
  v8f acc = {};

  for (int half = 0; half < 2; ++half) {
    __syncthreads();
    if (wave == 0) {
      // WfT rows are 256 K wide; pull a 128-wide K-half: stride 256, width 128
      tdm_load_2d_bf16((unsigned)(unsigned long long)&Wt[0][0], WfTg + half * 128,
                       128, OUT, 256, 128, OUT);
    }
    const float* srcp = (half == 0) ? h_gat : edge_agg;
    for (int i = tid; i < 16 * 128; i += 256) {
      int r = i >> 7, k = i & 127;
      long n = n0 + r;
      float v = 0.0f;
      if (n < N) {
        v = srcp[n * OUT + k];
        if (half == 1) v /= fmaxf(cnt[n], 1.0f);
      }
      As[r][k] = (__bf16)v;
    }
    if (wave == 0) __builtin_amdgcn_s_wait_tensorcnt(0);
    __syncthreads();
#pragma unroll
    for (int k0 = 0; k0 < 128; k0 += 32) {
      v16bf a = load_frag_a(&As[lane & 15][k0], lane);
      v16bf b = load_frag_b(&Wt[ct * 16 + (lane & 15)][k0], lane);
      acc = wmma_bf16(a, b, acc);
    }
  }

  int colg = ct * 16 + (lane & 15);
  float bv = bf_[colg];
#pragma unroll
  for (int j = 0; j < 8; ++j)
    lnbuf[j + ((lane >> 4) << 3)][colg] = acc[j] + bv;
  __syncthreads();

  int nr = tid >> 4, li = tid & 15;
  float s1 = 0.0f;
#pragma unroll
  for (int j = 0; j < 8; ++j) s1 += lnbuf[nr][li + 16 * j];
#pragma unroll
  for (int m = 1; m < 16; m <<= 1) s1 += __shfl_xor(s1, m, 16);
  float mean = s1 * (1.0f / 128.0f);
  float s2 = 0.0f;
#pragma unroll
  for (int j = 0; j < 8; ++j) {
    float dq = lnbuf[nr][li + 16 * j] - mean;
    s2 += dq * dq;
  }
#pragma unroll
  for (int m = 1; m < 16; m <<= 1) s2 += __shfl_xor(s2, m, 16);
  float rstd = rsqrtf(s2 * (1.0f / 128.0f) + 1e-5f);

  long n = n0 + nr;
  if (n < N) {
#pragma unroll
    for (int j = 0; j < 8; ++j) {
      int c = li + 16 * j;
      float y = (lnbuf[nr][c] - mean) * rstd * g[c] + bb[c];
      y = y > 0.0f ? y : expm1f(y);
      out[n * OUT + c] = y;
    }
  }
}

// ---------------- host launch ----------------

extern "C" void kernel_launch(void* const* d_in, const int* in_sizes, int n_in,
                              void* d_out, int out_size, void* d_ws, size_t ws_size,
                              hipStream_t stream) {
  const float* node_feat = (const float*)d_in[0];
  const float* edge_feat = (const float*)d_in[1];
  const float* W_src     = (const float*)d_in[2];
  const float* b_src     = (const float*)d_in[3];
  const float* W_dst     = (const float*)d_in[4];
  const float* b_dst     = (const float*)d_in[5];
  const float* attn      = (const float*)d_in[6];
  const float* We        = (const float*)d_in[7];
  const float* be        = (const float*)d_in[8];
  const float* ln1_g     = (const float*)d_in[9];
  const float* ln1_b     = (const float*)d_in[10];
  const float* Wf        = (const float*)d_in[11];
  const float* bf_       = (const float*)d_in[12];
  const float* ln2_g     = (const float*)d_in[13];
  const float* ln2_b     = (const float*)d_in[14];
  const int*   src       = (const int*)d_in[15];
  const int*   dst       = (const int*)d_in[16];

  const long N = in_sizes[0] / IN;
  const long E = in_sizes[15];
  const long EH = E * NH;

  // workspace layout
  float* ws        = (float*)d_ws;
  float* feat_src  = ws;                        // N*128
  float* feat_dst  = feat_src + N * OUT;        // N*128
  float* score     = feat_dst + N * OUT;        // E*4 (reused as ex)
  unsigned* smax   = (unsigned*)(score + EH);   // N*4
  float* denom     = (float*)(smax + N * NH);   // N*4
  float* h_gat     = denom + N * NH;            // N*128
  float* edge_agg  = h_gat + N * OUT;           // N*128
  float* cnt       = edge_agg + N * OUT;        // N
  __bf16* node_bf  = (__bf16*)(cnt + N);        // N*128 bf16
  __bf16* edge_bf  = node_bf + N * IN;          // E*32 bf16
  __bf16* WsrcT    = edge_bf + E * ED;          // 128*128 bf16 (transposed)
  __bf16* WdstT    = WsrcT + IN * OUT;          // 128*128
  __bf16* WeT      = WdstT + IN * OUT;          // 128*32
  __bf16* WfT      = WeT + (long)OUT * ED;      // 128*256

  float* out       = (float*)d_out;             // N*128
  float* out_alpha = out + N * OUT;             // E*4

  // precision prep: bf16 copies (activations row-major, weights transposed)
  long nn = N * IN;
  to_bf16_kernel<<<(unsigned)((nn + 255) / 256), 256, 0, stream>>>(node_feat, node_bf, nn);
  long ne = E * ED;
  to_bf16_kernel<<<(unsigned)((ne + 255) / 256), 256, 0, stream>>>(edge_feat, edge_bf, ne);
  transpose_bf16_kernel<<<(IN * OUT + 255) / 256, 256, 0, stream>>>(W_src, WsrcT, IN, OUT);
  transpose_bf16_kernel<<<(IN * OUT + 255) / 256, 256, 0, stream>>>(W_dst, WdstT, IN, OUT);
  transpose_bf16_kernel<<<(ED * OUT + 255) / 256, 256, 0, stream>>>(We, WeT, ED, OUT);
  transpose_bf16_kernel<<<(2 * OUT * OUT + 255) / 256, 256, 0, stream>>>(Wf, WfT, 2 * OUT, OUT);

  // zero accumulators (smax..cnt contiguous)
  long zn = N * (NH + NH + OUT + OUT + 1);
  zero_kernel<<<(unsigned)((zn + 255) / 256), 256, 0, stream>>>((float*)smax, zn);

  node_linear_kernel<<<(unsigned)((N + 31) / 32), 256, 0, stream>>>(
      node_bf, WsrcT, b_src, WdstT, b_dst, feat_src, feat_dst, (int)N);

  edge_score_kernel<<<(unsigned)((EH + 7) / 8), 256, 0, stream>>>(
      feat_src, feat_dst, attn, src, dst, score, smax, EH);

  edge_exp_kernel<<<(unsigned)((EH + 255) / 256), 256, 0, stream>>>(
      dst, smax, score, denom, EH);

  edge_aggregate_kernel<<<(unsigned)((E + 7) / 8), 256, 0, stream>>>(
      feat_src, src, dst, score, denom, h_gat, out_alpha, E);

  edge_emb_kernel<<<(unsigned)((E + 15) / 16), 256, 0, stream>>>(
      edge_bf, WeT, be, ln1_g, ln1_b, dst, edge_agg, cnt, E);

  final_kernel<<<(unsigned)((N + 15) / 16), 256, 0, stream>>>(
      h_gat, edge_agg, cnt, WfT, bf_, ln2_g, ln2_b, out, (int)N);
}
